// GatedDeltaNet_58334245814487
// MI455X (gfx1250) — matrix operand order repaired and checked by reading, not compiled, not run on hardware
//
#include <hip/hip_runtime.h>
#include <hip/hip_bf16.h>

// ---------------------------------------------------------------------------
// Types for CDNA5 WMMA (wave32): v16 bf16 A/B fragments, v8 f32 accumulator.
// ---------------------------------------------------------------------------
typedef __attribute__((ext_vector_type(16))) __bf16 bf16x16;
typedef __attribute__((ext_vector_type(8)))  __bf16 bf16x8;
typedef __attribute__((ext_vector_type(8)))  float  f32x8;

__device__ __forceinline__ unsigned lane_id() { return threadIdx.x & 31u; }

__device__ __forceinline__ __bf16 to_bf(float f) {
  unsigned u = __float_as_uint(f);
  unsigned r = (u + 0x7FFFu + ((u >> 16) & 1u)) >> 16;
  unsigned short s = (unsigned short)r;
  return __builtin_bit_cast(__bf16, s);
}
__device__ __forceinline__ float bf2f(__bf16 b) {
  unsigned short s = __builtin_bit_cast(unsigned short, b);
  return __uint_as_float(((unsigned)s) << 16);
}
__device__ __forceinline__ f32x8 zero8() {
  f32x8 z;
#pragma unroll
  for (int i = 0; i < 8; ++i) z[i] = 0.f;
  return z;
}

// A-operand 16x32 bf16 (ISA 7.12.2): lane = h*16+m; elems 0..7 -> K=8h+0..7,
// elems 8..15 -> K=16+8h+0..7. Rows are k-contiguous in memory.
__device__ __forceinline__ bf16x16 load_a_frag(const __bf16* A, int lda, int k0) {
  unsigned l = lane_id(), m = l & 15u, h = l >> 4;
  const __bf16* p = A + (size_t)m * lda + k0 + h * 8;
  bf16x8 lo = *(const bf16x8*)p;
  bf16x8 hi = *(const bf16x8*)(p + 16);
  bf16x16 f;
#pragma unroll
  for (int i = 0; i < 8; ++i) { f[i] = lo[i]; f[i + 8] = hi[i]; }
  return f;
}

// B-operand 32x16 bf16 from a row-major "B-transposed" buffer BT[n][k]:
// lane = h*16+n; elems 0..7 -> K=8h+0..7, elems 8..15 -> K=16+8h+0..7 (BT row n).
__device__ __forceinline__ bf16x16 load_b_frag(const __bf16* BT, int ldb, int k0) {
  unsigned l = lane_id(), n = l & 15u, h = l >> 4;
  const __bf16* p = BT + (size_t)n * ldb + k0 + h * 8;
  bf16x8 lo = *(const bf16x8*)p;
  bf16x8 hi = *(const bf16x8*)(p + 16);
  bf16x16 f;
#pragma unroll
  for (int i = 0; i < 8; ++i) { f[i] = lo[i]; f[i + 8] = hi[i]; }
  return f;
}

__device__ __forceinline__ f32x8 wmma_bf16(bf16x16 a, bf16x16 b, f32x8 c) {
  return __builtin_amdgcn_wmma_f32_16x16x32_bf16(false, a, false, b, (short)0, c,
                                                 false, false);
}

// K-loop over one 16-row A strip and NS 16-col B strips.
template <int NS>
__device__ __forceinline__ void mma_k(f32x8 acc[NS], const __bf16* Arow, int lda,
                                      const __bf16* BT0, int ldb, int K) {
  for (int k0 = 0; k0 < K; k0 += 32) {
    bf16x16 a = load_a_frag(Arow, lda, k0);
#pragma unroll
    for (int s = 0; s < NS; ++s) {
      bf16x16 b = load_b_frag(BT0 + (size_t)(s * 16) * ldb, ldb, k0);
      acc[s] = wmma_bf16(a, b, acc[s]);
    }
  }
}

// D layout (f32 16x16): elem v of lane (h*16+n) -> row v+8h, col n.
__device__ __forceinline__ void store_d(__bf16* C, int ldc, f32x8 d) {
  unsigned l = lane_id(), n = l & 15u, h = l >> 4;
#pragma unroll
  for (int v = 0; v < 8; ++v) C[(size_t)(v + 8 * h) * ldc + n] = to_bf(d[v]);
}
__device__ __forceinline__ void store_d(float* C, int ldc, f32x8 d) {
  unsigned l = lane_id(), n = l & 15u, h = l >> 4;
#pragma unroll
  for (int v = 0; v < 8; ++v) C[(size_t)(v + 8 * h) * ldc + n] = d[v];
}
__device__ __forceinline__ f32x8 load_d_f32(const float* C, int ldc) {
  unsigned l = lane_id(), n = l & 15u, h = l >> 4;
  f32x8 d;
#pragma unroll
  for (int v = 0; v < 8; ++v) d[v] = C[(size_t)(v + 8 * h) * ldc + n];
  return d;
}
// Transposed store: C[col * ldcT + row], C pre-offset to (col0,row0).
__device__ __forceinline__ void store_d_bf16_T(__bf16* C, int ldcT, f32x8 d) {
  unsigned l = lane_id(), n = l & 15u, h = l >> 4;
#pragma unroll
  for (int v = 0; v < 8; ++v) C[(size_t)n * ldcT + (v + 8 * h)] = to_bf(d[v]);
}

// ---------------------------------------------------------------------------
// Elementwise / setup kernels
// ---------------------------------------------------------------------------
__global__ void k_f32_to_bf16(const float* __restrict__ src, __bf16* __restrict__ dst,
                              int n) {
  for (int i = blockIdx.x * blockDim.x + threadIdx.x; i < n;
       i += gridDim.x * blockDim.x)
    dst[i] = to_bf(src[i]);
}

// W[K][N] f32 -> WT[N][K] bf16 (coalesced writes).
__global__ void k_w_transpose(const float* __restrict__ W, __bf16* __restrict__ WT,
                              int K, int N) {
  int total = K * N;
  for (int e = blockIdx.x * blockDim.x + threadIdx.x; e < total;
       e += gridDim.x * blockDim.x) {
    int k = e % K, n = e / K;
    WT[e] = to_bf(W[(size_t)k * N + n]);
  }
}

// Row-wise: L2-normalize k (eps=1e-12) and scale q by DK^-0.5 = 1/32.
__global__ void k_norm(__bf16* __restrict__ qbf, __bf16* __restrict__ kbf) {
  __shared__ float red[256];
  size_t row = blockIdx.x;
  __bf16* kr = kbf + row * 1024;
  __bf16* qr = qbf + row * 1024;
  int t = threadIdx.x;
  float ss = 0.f;
  for (int i = t; i < 1024; i += 256) { float f = bf2f(kr[i]); ss += f * f; }
  red[t] = ss;
  __syncthreads();
  for (int s2 = 128; s2 > 0; s2 >>= 1) {
    if (t < s2) red[t] += red[t + s2];
    __syncthreads();
  }
  float r = 1.f / fmaxf(sqrtf(red[0]), 1e-12f);
  for (int i = t; i < 1024; i += 256) {
    kr[i] = to_bf(bf2f(kr[i]) * r);
    qr[i] = to_bf(bf2f(qr[i]) * 0.03125f);
  }
}

// ---------------------------------------------------------------------------
// High-intensity GEMM: C[M,N] = A[M,K] * BT[N,K]^T.
// Block = 256 thr (8 waves), tile 128(M) x 128(N); wave -> 32x64 strip:
// per k-step 6 fragments (12x b128) feed 8 WMMAs, A fragments reused 4x, B fragments 2x.
// ---------------------------------------------------------------------------
template <typename OutT>
__global__ __launch_bounds__(256) void k_gemm_big(
    const __bf16* __restrict__ A, int lda, const __bf16* __restrict__ BT, int ldb,
    OutT* __restrict__ C, int ldc, int K) {
  int w = threadIdx.x >> 5;
  int m0 = blockIdx.y * 128 + (w & 3) * 32;
  int n0 = blockIdx.x * 128 + (w >> 2) * 64;
  f32x8 acc[2][4];
#pragma unroll
  for (int mi = 0; mi < 2; ++mi)
#pragma unroll
    for (int s = 0; s < 4; ++s) acc[mi][s] = zero8();
  const __bf16* A0 = A + (size_t)m0 * lda;
  const __bf16* A1 = A0 + (size_t)16 * lda;
  const __bf16* B0 = BT + (size_t)n0 * ldb;
  for (int k0 = 0; k0 < K; k0 += 32) {
    bf16x16 a0 = load_a_frag(A0, lda, k0);
    bf16x16 a1 = load_a_frag(A1, lda, k0);
    bf16x16 b0 = load_b_frag(B0, ldb, k0);
    bf16x16 b1 = load_b_frag(B0 + (size_t)16 * ldb, ldb, k0);
    bf16x16 b2 = load_b_frag(B0 + (size_t)32 * ldb, ldb, k0);
    bf16x16 b3 = load_b_frag(B0 + (size_t)48 * ldb, ldb, k0);
    acc[0][0] = wmma_bf16(a0, b0, acc[0][0]);
    acc[1][0] = wmma_bf16(a1, b0, acc[1][0]);
    acc[0][1] = wmma_bf16(a0, b1, acc[0][1]);
    acc[1][1] = wmma_bf16(a1, b1, acc[1][1]);
    acc[0][2] = wmma_bf16(a0, b2, acc[0][2]);
    acc[1][2] = wmma_bf16(a1, b2, acc[1][2]);
    acc[0][3] = wmma_bf16(a0, b3, acc[0][3]);
    acc[1][3] = wmma_bf16(a1, b3, acc[1][3]);
  }
#pragma unroll
  for (int mi = 0; mi < 2; ++mi)
#pragma unroll
    for (int s = 0; s < 4; ++s)
      store_d(C + (size_t)(m0 + mi * 16) * ldc + n0 + s * 16, ldc, acc[mi][s]);
}

// ---------------------------------------------------------------------------
// Per-chunk kernels (chunkwise-parallel delta rule, L = 64)
// ---------------------------------------------------------------------------

// blockIdx.x==1: Amat = tril(Kc Kc^T, -1) f32.  blockIdx.x==0: G = tril(Qc Kc^T) bf16.
__global__ __launch_bounds__(256) void k_attn(const __bf16* __restrict__ qbf,
                                              const __bf16* __restrict__ kbf, int T,
                                              int chunk, float* __restrict__ Amat,
                                              __bf16* __restrict__ Gbf) {
  int b = blockIdx.y;
  int which = blockIdx.x;  // 1 -> A (strict lower), 0 -> G (incl. diag)
  const __bf16* Kc = kbf + ((size_t)b * T + (size_t)chunk * 64) * 1024;
  const __bf16* Lm = which ? Kc : (qbf + ((size_t)b * T + (size_t)chunk * 64) * 1024);
  int w = threadIdx.x >> 5;
  int m0 = (w & 3) * 16;
  int n0 = (w >> 2) * 32;
  f32x8 acc[2];
#pragma unroll
  for (int s = 0; s < 2; ++s) acc[s] = zero8();
  mma_k<2>(acc, Lm + (size_t)m0 * 1024, 1024, Kc + (size_t)n0 * 1024, 1024, 1024);
  unsigned l = lane_id(), n = l & 15u, h = l >> 4;
#pragma unroll
  for (int s = 0; s < 2; ++s) {
#pragma unroll
    for (int v = 0; v < 8; ++v) {
      int row = m0 + v + 8 * h;
      int col = n0 + s * 16 + n;
      float val = acc[s][v];
      if (which) {
        Amat[((size_t)b * 64 + row) * 64 + col] = (col < row) ? val : 0.f;
      } else {
        Gbf[((size_t)b * 64 + row) * 64 + col] = to_bf((col <= row) ? val : 0.f);
      }
    }
  }
}

// Kc[64][1024] -> KcT[1024][64] bf16
__global__ void k_kct(const __bf16* __restrict__ kbf, int T, int chunk,
                      __bf16* __restrict__ KcT) {
  int b = blockIdx.y;
  const __bf16* Kc = kbf + ((size_t)b * T + (size_t)chunk * 64) * 1024;
  __bf16* Tt = KcT + (size_t)b * 1024 * 64;
  for (int e = blockIdx.x * blockDim.x + threadIdx.x; e < 64 * 1024;
       e += gridDim.x * blockDim.x) {
    int t = e & 63, d = e >> 6;
    Tt[e] = Kc[(size_t)t * 1024 + d];
  }
}

// Tinv = (I + A)^{-1}, unit lower-triangular, forward substitution in LDS.
__global__ __launch_bounds__(64) void k_solve(const float* __restrict__ Amat,
                                              __bf16* __restrict__ Tinv) {
  __shared__ float X[64][64];
  __shared__ float Arow[64];
  int b = blockIdx.x;
  const float* A = Amat + (size_t)b * 64 * 64;
  int j = threadIdx.x;
  for (int i = 0; i < 64; ++i) {
    Arow[j] = A[(size_t)i * 64 + j];
    __syncthreads();
    float val = (i == j) ? 1.f : 0.f;
    for (int p = 0; p < i; ++p) val -= Arow[p] * X[p][j];
    X[i][j] = (j <= i) ? val : 0.f;
    __syncthreads();
  }
  __bf16* Tb = Tinv + (size_t)b * 64 * 64;
  for (int i = 0; i < 64; ++i) Tb[(size_t)i * 64 + j] = to_bf(X[i][j]);
}

// RHS^T[v][t] = V[t][v] - (Kc @ S^T)[t][v]   (S mirror is bf16, rows k-contig)
__global__ __launch_bounds__(256) void k_rhs(const __bf16* __restrict__ kbf,
                                             const __bf16* __restrict__ vbf,
                                             const __bf16* __restrict__ Sbf, int T,
                                             int chunk, __bf16* __restrict__ RHSt) {
  int b = blockIdx.y;
  const __bf16* Kc = kbf + ((size_t)b * T + (size_t)chunk * 64) * 1024;
  const __bf16* Vc = vbf + ((size_t)b * T + (size_t)chunk * 64) * 1024;
  const __bf16* S = Sbf + (size_t)b * 1024 * 1024;
  __bf16* R = RHSt + (size_t)b * 1024 * 64;
  int w = threadIdx.x >> 5;
  int m0 = (w & 3) * 16;
  int n0 = blockIdx.x * 128 + (w >> 2) * 64;
  f32x8 acc[4];
#pragma unroll
  for (int s = 0; s < 4; ++s) acc[s] = zero8();
  mma_k<4>(acc, Kc + (size_t)m0 * 1024, 1024, S + (size_t)n0 * 1024, 1024, 1024);
  unsigned l = lane_id(), n = l & 15u, h = l >> 4;
#pragma unroll
  for (int s = 0; s < 4; ++s) {
#pragma unroll
    for (int v = 0; v < 8; ++v) {
      int row = m0 + v + 8 * h;
      int col = n0 + s * 16 + n;
      float val = bf2f(Vc[(size_t)row * 1024 + col]) - acc[s][v];
      R[(size_t)col * 64 + row] = to_bf(val);
    }
  }
}

// U^T[v][t] = (Tinv @ RHS)[t][v]
__global__ __launch_bounds__(256) void k_u(const __bf16* __restrict__ Tinv,
                                           const __bf16* __restrict__ RHSt,
                                           __bf16* __restrict__ Ut) {
  int b = blockIdx.y;
  const __bf16* Tb = Tinv + (size_t)b * 4096;
  const __bf16* R = RHSt + (size_t)b * 65536;
  __bf16* U = Ut + (size_t)b * 65536;
  int w = threadIdx.x >> 5;
  int m0 = (w & 3) * 16;
  int n0 = blockIdx.x * 128 + (w >> 2) * 64;
  f32x8 acc[4];
#pragma unroll
  for (int s = 0; s < 4; ++s) acc[s] = zero8();
  mma_k<4>(acc, Tb + (size_t)m0 * 64, 64, R + (size_t)n0 * 64, 64, 64);
#pragma unroll
  for (int s = 0; s < 4; ++s)
    store_d_bf16_T(U + (size_t)(n0 + s * 16) * 64 + m0, 64, acc[s]);
}

// O[t][v] = (Qc @ S^T)[t][v] + (G @ U)[t][v]
__global__ __launch_bounds__(256) void k_out(const __bf16* __restrict__ qbf,
                                             const __bf16* __restrict__ Sbf,
                                             const __bf16* __restrict__ Gbf,
                                             const __bf16* __restrict__ Ut, int T,
                                             int chunk, __bf16* __restrict__ Obf) {
  int b = blockIdx.y;
  const __bf16* Qc = qbf + ((size_t)b * T + (size_t)chunk * 64) * 1024;
  const __bf16* S = Sbf + (size_t)b * 1024 * 1024;
  const __bf16* G = Gbf + (size_t)b * 4096;
  const __bf16* U = Ut + (size_t)b * 65536;
  __bf16* O = Obf + ((size_t)b * T + (size_t)chunk * 64) * 1024;
  int w = threadIdx.x >> 5;
  int m0 = (w & 3) * 16;
  int n0 = blockIdx.x * 128 + (w >> 2) * 64;
  f32x8 acc[4];
#pragma unroll
  for (int s = 0; s < 4; ++s) acc[s] = zero8();
  mma_k<4>(acc, Qc + (size_t)m0 * 1024, 1024, S + (size_t)n0 * 1024, 1024, 1024);
  mma_k<4>(acc, G + (size_t)m0 * 64, 64, U + (size_t)n0 * 64, 64, 64);
#pragma unroll
  for (int s = 0; s < 4; ++s)
    store_d(O + (size_t)m0 * 1024 + n0 + s * 16, 1024, acc[s]);
}

// S[v][d] += (U^T Kc)[v][d]; refresh bf16 mirror.  128x128 block tile,
// 32x64 wave tile, K = 64 (2 k-steps).
__global__ __launch_bounds__(256) void k_state(const __bf16* __restrict__ Ut,
                                               const __bf16* __restrict__ KcT,
                                               float* __restrict__ Sf,
                                               __bf16* __restrict__ Sbf) {
  int b = blockIdx.z;
  const __bf16* U = Ut + (size_t)b * 65536;
  const __bf16* Kt = KcT + (size_t)b * 65536;
  float* S = Sf + (size_t)b * 1048576;
  __bf16* Sb = Sbf + (size_t)b * 1048576;
  int w = threadIdx.x >> 5;
  int m0 = blockIdx.y * 128 + (w & 3) * 32;   // v
  int n0 = blockIdx.x * 128 + (w >> 2) * 64;  // d
  f32x8 acc[2][4];
#pragma unroll
  for (int mi = 0; mi < 2; ++mi)
#pragma unroll
    for (int s = 0; s < 4; ++s)
      acc[mi][s] = load_d_f32(S + (size_t)(m0 + mi * 16) * 1024 + n0 + s * 16, 1024);
  const __bf16* U0 = U + (size_t)m0 * 64;
  const __bf16* U1 = U0 + (size_t)16 * 64;
  const __bf16* K0 = Kt + (size_t)n0 * 64;
  for (int k0 = 0; k0 < 64; k0 += 32) {
    bf16x16 a0 = load_a_frag(U0, 64, k0);
    bf16x16 a1 = load_a_frag(U1, 64, k0);
    bf16x16 b0 = load_b_frag(K0, 64, k0);
    bf16x16 b1 = load_b_frag(K0 + (size_t)16 * 64, 64, k0);
    bf16x16 b2 = load_b_frag(K0 + (size_t)32 * 64, 64, k0);
    bf16x16 b3 = load_b_frag(K0 + (size_t)48 * 64, 64, k0);
    acc[0][0] = wmma_bf16(a0, b0, acc[0][0]);
    acc[1][0] = wmma_bf16(a1, b0, acc[1][0]);
    acc[0][1] = wmma_bf16(a0, b1, acc[0][1]);
    acc[1][1] = wmma_bf16(a1, b1, acc[1][1]);
    acc[0][2] = wmma_bf16(a0, b2, acc[0][2]);
    acc[1][2] = wmma_bf16(a1, b2, acc[1][2]);
    acc[0][3] = wmma_bf16(a0, b3, acc[0][3]);
    acc[1][3] = wmma_bf16(a1, b3, acc[1][3]);
  }
#pragma unroll
  for (int mi = 0; mi < 2; ++mi)
#pragma unroll
    for (int s = 0; s < 4; ++s) {
      size_t o = (size_t)(m0 + mi * 16) * 1024 + n0 + s * 16;
      store_d(S + o, 1024, acc[mi][s]);
      store_d(Sb + o, 1024, acc[mi][s]);
    }
}

// ---------------------------------------------------------------------------
// Host launcher
// ---------------------------------------------------------------------------
extern "C" void kernel_launch(void* const* d_in, const int* in_sizes, int n_in,
                              void* d_out, int out_size, void* d_ws, size_t ws_size,
                              hipStream_t stream) {
  const int B = 4, T = 2048, D = 1024;
  const int L = 64, NC = T / L;  // 32 chunks
  const int BT = B * T;          // 8192

  const float* x  = (const float*)d_in[0];
  const float* Wq = (const float*)d_in[1];
  const float* Wk = (const float*)d_in[2];
  const float* Wv = (const float*)d_in[3];
  const float* Wo = (const float*)d_in[4];
  float* out = (float*)d_out;

  size_t off = 0;
  auto alloc = [&](size_t bytes) {
    size_t o = off;
    off += (bytes + 255) & ~(size_t)255;
    return o;
  };
  char* ws = (char*)d_ws;
  const size_t big = (size_t)BT * D * sizeof(__bf16);   // 16 MB
  const size_t wsz = (size_t)D * D * sizeof(__bf16);    // 2 MB

  __bf16* xbf  = (__bf16*)(ws + alloc(big));
  __bf16* qbf  = (__bf16*)(ws + alloc(big));
  __bf16* kbf  = (__bf16*)(ws + alloc(big));
  __bf16* vbf  = (__bf16*)(ws + alloc(big));
  __bf16* obf  = (__bf16*)(ws + alloc(big));
  __bf16* WqT  = (__bf16*)(ws + alloc(wsz));
  __bf16* WkT  = (__bf16*)(ws + alloc(wsz));
  __bf16* WvT  = (__bf16*)(ws + alloc(wsz));
  __bf16* WoT  = (__bf16*)(ws + alloc(wsz));
  float*  S    = (float*) (ws + alloc((size_t)B * D * D * sizeof(float)));   // 16 MB
  __bf16* Sbf  = (__bf16*)(ws + alloc((size_t)B * D * D * sizeof(__bf16))); // 8 MB
  float*  Amat = (float*) (ws + alloc((size_t)B * L * L * sizeof(float)));
  __bf16* Gbf  = (__bf16*)(ws + alloc((size_t)B * L * L * sizeof(__bf16)));
  __bf16* Tinv = (__bf16*)(ws + alloc((size_t)B * L * L * sizeof(__bf16)));
  __bf16* RHSt = (__bf16*)(ws + alloc((size_t)B * D * L * sizeof(__bf16)));
  __bf16* Ut   = (__bf16*)(ws + alloc((size_t)B * D * L * sizeof(__bf16)));
  __bf16* KcT  = (__bf16*)(ws + alloc((size_t)B * D * L * sizeof(__bf16)));
  (void)ws_size; (void)in_sizes; (void)n_in; (void)out_size;

  // 1) fp32 -> bf16 conversions / weight transposes
  k_f32_to_bf16<<<4096, 256, 0, stream>>>(x, xbf, BT * D);
  k_w_transpose<<<1024, 256, 0, stream>>>(Wq, WqT, D, D);
  k_w_transpose<<<1024, 256, 0, stream>>>(Wk, WkT, D, D);
  k_w_transpose<<<1024, 256, 0, stream>>>(Wv, WvT, D, D);
  k_w_transpose<<<1024, 256, 0, stream>>>(Wo, WoT, D, D);

  // 2) Projections q/k/v = x @ W (WMMA bf16, 128x128 tiles)
  dim3 gBig(D / 128, BT / 128);
  k_gemm_big<__bf16><<<gBig, 256, 0, stream>>>(xbf, D, WqT, D, qbf, D, D);
  k_gemm_big<__bf16><<<gBig, 256, 0, stream>>>(xbf, D, WkT, D, kbf, D, D);
  k_gemm_big<__bf16><<<gBig, 256, 0, stream>>>(xbf, D, WvT, D, vbf, D, D);

  // 3) normalize k rows, scale q rows
  k_norm<<<BT, 256, 0, stream>>>(qbf, kbf);

  // 4) init state S = 0 (f32 + bf16 mirror)
  hipMemsetAsync(S, 0, (size_t)B * D * D * sizeof(float), stream);
  hipMemsetAsync(Sbf, 0, (size_t)B * D * D * sizeof(__bf16), stream);

  // 5) chunkwise-parallel delta-rule scan
  for (int c = 0; c < NC; ++c) {
    k_attn <<<dim3(2, B),       256, 0, stream>>>(qbf, kbf, T, c, Amat, Gbf);
    k_kct  <<<dim3(64, B),      256, 0, stream>>>(kbf, T, c, KcT);
    k_solve<<<B,                 64, 0, stream>>>(Amat, Tinv);
    k_rhs  <<<dim3(D / 128, B), 256, 0, stream>>>(kbf, vbf, Sbf, T, c, RHSt);
    k_u    <<<dim3(D / 128, B), 256, 0, stream>>>(Tinv, RHSt, Ut);
    k_out  <<<dim3(D / 128, B), 256, 0, stream>>>(qbf, Sbf, Gbf, Ut, T, c, obf);
    k_state<<<dim3(D / 128, D / 128, B), 256, 0, stream>>>(Ut, KcT, S, Sbf);
  }

  // 6) final projection: out = O @ Wo (f32 output)
  k_gemm_big<float><<<gBig, 256, 0, stream>>>(obf, D, WoT, D, out, D, D);
}
